// CrossAttention_57844619542886
// MI455X (gfx1250) — matrix operand order repaired and checked by reading, not compiled, not run on hardware
//
#include <hip/hip_runtime.h>

// ---------------- problem constants ----------------
#define Nn   8
#define Cc   256
#define OUTc 64
#define HWs  2304   // 48*48

typedef unsigned short u16;
typedef __attribute__((ext_vector_type(16))) __bf16 v16bf;
typedef __attribute__((ext_vector_type(8)))  float  v8f;
typedef __attribute__((ext_vector_type(4)))  unsigned int v4u;

union Frag {
  v16bf v;
  v4u   q[2];
  u16   u[16];
};

__device__ __forceinline__ u16 f2bf(float f) {
  union { float f; unsigned u; } a; a.f = f;
  unsigned r = a.u + 0x7fffu + ((a.u >> 16) & 1u);
  return (u16)(r >> 16);
}

// B-operand fragment (16-bit, 32x16): lane holds column N=lane%16,
// element e <-> k = (e<8 ? e : e+8) + 8*(lane/16): two contiguous 8-runs.
__device__ __forceinline__ v16bf load_b(const u16* rowp, int kc, int h) {
  Frag f;
  f.q[0] = *(const v4u*)(rowp + kc + 8 * h);
  f.q[1] = *(const v4u*)(rowp + kc + 16 + 8 * h);
  return f.v;
}

// A-operand fragment (16-bit, 16x32): lane holds row M=lane%16,
// element e <-> k = (e<8 ? e : e+8) + 8*(lane/16): two contiguous 8-runs.
__device__ __forceinline__ v16bf load_a(const u16* rowp, int kc, int h) {
  Frag f;
  f.q[0] = *(const v4u*)(rowp + kc + 8 * h);
  f.q[1] = *(const v4u*)(rowp + kc + 16 + 8 * h);
  return f.v;
}

__device__ __forceinline__ v8f wmma_bf16(v16bf a, v16bf b, v8f c) {
  return __builtin_amdgcn_wmma_f32_16x16x32_bf16(
      /*neg_a=*/false, a, /*neg_b=*/false, b,
      /*c_mod=*/(short)0, c, /*reuse_a=*/false, /*reuse_b=*/false);
}

// ---------------- kernel 0a: fp32 [n][c][s] -> bf16 [n][s][c] (x and mask) ----------------
__global__ void k_transpose_bf16(const float* __restrict__ x, const float* __restrict__ mask,
                                 u16* __restrict__ xT, u16* __restrict__ mT) {
  __shared__ float tile[32][33];
  int which = blockIdx.z >> 3;          // 0: x, 1: mask
  int n = blockIdx.z & 7;
  int c0 = blockIdx.y * 32;
  int s0 = blockIdx.x * 32;
  const float* src = (which ? mask : x) + (size_t)n * Cc * HWs;
  u16* dst = (which ? mT : xT) + (size_t)n * HWs * Cc;
  int tx = threadIdx.x, ty = threadIdx.y;  // 32 x 8
#pragma unroll
  for (int k = 0; k < 32; k += 8)
    tile[ty + k][tx] = src[(size_t)(c0 + ty + k) * HWs + s0 + tx];
  __syncthreads();
#pragma unroll
  for (int k = 0; k < 32; k += 8)
    dst[(size_t)(s0 + ty + k) * Cc + c0 + tx] = f2bf(tile[tx][ty + k]);
}

// ---------------- kernel 0b: pack weights to bf16 ----------------
__global__ void k_pack_weights(const float* __restrict__ Wf, const float* __restrict__ Wg,
                               const float* __restrict__ Wh, const float* __restrict__ Wm,
                               const float* __restrict__ Wv,
                               u16* __restrict__ Wb, u16* __restrict__ Wvb) {
  int t = blockIdx.x * 256 + threadIdx.x;
  if (t < 4 * OUTc * Cc) {
    int p = t / (OUTc * Cc);
    int r = t % (OUTc * Cc);
    const float* s = (p == 0) ? Wf : (p == 1) ? Wg : (p == 2) ? Wh : Wm;
    Wb[t] = f2bf(s[r]);
  }
  if (t < Cc * OUTc) Wvb[t] = f2bf(Wv[t]);
}

// ---------------- kernel 1: projections via WMMA ----------------
// PT[n][s][o] = sum_c srcT[n][s][c] * W[o][c] + b[o]
// p=0 -> PfT bf16 [s][64], p=1 -> PgT bf16 [s][64],
// p=2 -> Ph  bf16 [64][s] (from mask), p=3 -> PmT fp32 [s][64]
__global__ void __launch_bounds__(128) k_proj(
    const u16* __restrict__ xT, const u16* __restrict__ mT, const u16* __restrict__ Wb,
    const float* __restrict__ bf_, const float* __restrict__ bg_,
    const float* __restrict__ bh_, const float* __restrict__ bm_,
    u16* __restrict__ PfT, u16* __restrict__ PgT, u16* __restrict__ Ph,
    float* __restrict__ PmT) {
  int p = blockIdx.y;
  int n = blockIdx.z;
  int lane = threadIdx.x & 31, wave = threadIdx.x >> 5;
  int h = lane >> 4, ln = lane & 15;
  int sb = blockIdx.x * 64 + wave * 16;

  const u16* src = ((p == 2) ? mT : xT) + (size_t)n * HWs * Cc;
  const u16* Wp = Wb + (size_t)p * OUTc * Cc;
  const u16* arow = src + (size_t)(sb + ln) * Cc;

  v8f acc[4] = {};
  for (int kc = 0; kc < Cc; kc += 32) {
    v16bf a = load_a(arow, kc, h);
#pragma unroll
    for (int ot = 0; ot < 4; ++ot) {
      v16bf b = load_b(Wp + (size_t)(ot * 16 + ln) * Cc, kc, h);
      acc[ot] = wmma_bf16(a, b, acc[ot]);
    }
  }

  const float* bias = (p == 0) ? bf_ : (p == 1) ? bg_ : (p == 2) ? bh_ : bm_;
  if (p == 2) {
    u16* dst = Ph + (size_t)n * OUTc * HWs;
#pragma unroll
    for (int ot = 0; ot < 4; ++ot) {
      int o = ot * 16 + ln;
      float bo = bias[o];
      Frag f;
#pragma unroll
      for (int v = 0; v < 8; ++v) f.u[v] = f2bf(acc[ot][v] + bo);
      // rows v are contiguous s -> one 16B store
      *(v4u*)(dst + (size_t)o * HWs + sb + 8 * h) = f.q[0];
    }
  } else if (p == 3) {
    float* dst = PmT + (size_t)n * HWs * OUTc;
#pragma unroll
    for (int ot = 0; ot < 4; ++ot) {
      int o = ot * 16 + ln;
      float bo = bias[o];
#pragma unroll
      for (int v = 0; v < 8; ++v)
        dst[(size_t)(sb + v + 8 * h) * OUTc + o] = acc[ot][v] + bo;
    }
  } else {
    u16* dst = ((p == 0) ? PfT : PgT) + (size_t)n * HWs * OUTc;
#pragma unroll
    for (int ot = 0; ot < 4; ++ot) {
      int o = ot * 16 + ln;
      float bo = bias[o];
#pragma unroll
      for (int v = 0; v < 8; ++v)
        dst[(size_t)(sb + v + 8 * h) * OUTc + o] = f2bf(acc[ot][v] + bo);
    }
  }
}

// ---------------- kernel 2: fused flash attention ----------------
// O[n][i][c] = sum_j softmax_j(Q[i]·K[j]) * V[c][j]   (online softmax over j)
__global__ void __launch_bounds__(128) k_attn(
    const u16* __restrict__ PfT, const u16* __restrict__ PgT,
    const u16* __restrict__ Ph, float* __restrict__ O) {
  __shared__ u16 ldsP[4][16][40];  // per-wave 16x32 P tile, padded rows
  int n = blockIdx.y;
  int lane = threadIdx.x & 31, wave = threadIdx.x >> 5;
  int h = lane >> 4, ln = lane & 15;
  int ib = blockIdx.x * 64 + wave * 16;

  const u16* Q = PfT + (size_t)n * HWs * OUTc;
  const u16* K = PgT + (size_t)n * HWs * OUTc;
  const u16* V = Ph + (size_t)n * OUTc * HWs;

  const u16* qrow = Q + (size_t)(ib + ln) * OUTc;
  v16bf q0 = load_a(qrow, 0, h);
  v16bf q1 = load_a(qrow, 32, h);

  v8f acc[4] = {};
  float m[8], l[8];
#pragma unroll
  for (int v = 0; v < 8; ++v) { m[v] = -3.0e38f; l[v] = 0.0f; }

  for (int j = 0; j < HWs; j += 32) {
    const u16* k0row = K + (size_t)(j + ln) * OUTc;
    const u16* k1row = K + (size_t)(j + 16 + ln) * OUTc;
    __builtin_prefetch(k0row + 32 * OUTc, 0, 1);  // next j-tile -> global_prefetch_b8

    v8f s0 = {}, s1 = {};
    s0 = wmma_bf16(q0, load_b(k0row, 0, h), s0);
    s0 = wmma_bf16(q1, load_b(k0row, 32, h), s0);
    s1 = wmma_bf16(q0, load_b(k1row, 0, h), s1);
    s1 = wmma_bf16(q1, load_b(k1row, 32, h), s1);

    float p0[8], p1[8], cr[8];
#pragma unroll
    for (int v = 0; v < 8; ++v) {
      float mx = fmaxf(s0[v], s1[v]);
#pragma unroll
      for (int d = 1; d < 16; d <<= 1) mx = fmaxf(mx, __shfl_xor(mx, d));
      float mn = fmaxf(m[v], mx);
      cr[v] = __expf(m[v] - mn);
      float e0 = __expf(s0[v] - mn);
      float e1 = __expf(s1[v] - mn);
      float rs = e0 + e1;
#pragma unroll
      for (int d = 1; d < 16; d <<= 1) rs += __shfl_xor(rs, d);
      l[v] = l[v] * cr[v] + rs;
      m[v] = mn;
      p0[v] = e0; p1[v] = e1;
    }
#pragma unroll
    for (int ct = 0; ct < 4; ++ct)
#pragma unroll
      for (int v = 0; v < 8; ++v) acc[ct][v] *= cr[v];

    // D-layout P -> LDS -> A-layout fragment (wave-private, LDS in-order)
#pragma unroll
    for (int v = 0; v < 8; ++v) {
      ldsP[wave][v + 8 * h][ln] = f2bf(p0[v]);
      ldsP[wave][v + 8 * h][16 + ln] = f2bf(p1[v]);
    }
    Frag ap;
    const u16* lp = &ldsP[wave][ln][0];
    ap.q[0] = *(const v4u*)(lp + 8 * h);
    ap.q[1] = *(const v4u*)(lp + 16 + 8 * h);

#pragma unroll
    for (int ct = 0; ct < 4; ++ct) {
      const u16* vrow = V + (size_t)(ct * 16 + ln) * HWs;
      acc[ct] = wmma_bf16(ap.v, load_b(vrow + j, 0, h), acc[ct]);
    }
  }

  float* Ob = O + (size_t)n * HWs * OUTc;
#pragma unroll
  for (int ct = 0; ct < 4; ++ct) {
    int c = ct * 16 + ln;
#pragma unroll
    for (int v = 0; v < 8; ++v) {
      int i = ib + v + 8 * h;
      Ob[(size_t)i * OUTc + c] = acc[ct][v] / l[v];
    }
  }
}

// ---------------- kernel 3: column softmax stats over i ----------------
__global__ void k_colstats(const float* __restrict__ O,
                           float* __restrict__ colmax, float* __restrict__ colsum) {
  __shared__ float sm[4][64], sl[4][64];
  int n = blockIdx.x;
  int t = threadIdx.x;
  int c = t & 63, g = t >> 6;
  const float* Ob = O + (size_t)n * HWs * OUTc;
  float m = -3.0e38f, l = 0.0f;
  for (int r = g; r < HWs; r += 4) {
    float v = Ob[(size_t)r * OUTc + c];
    float mn = fmaxf(m, v);
    l = l * __expf(m - mn) + __expf(v - mn);
    m = mn;
  }
  sm[g][c] = m; sl[g][c] = l;
  __syncthreads();
  if (g == 0) {
    float M = m, L = l;
#pragma unroll
    for (int k = 1; k < 4; ++k) {
      float mk = sm[k][c], lk = sl[k][c];
      float mn = fmaxf(M, mk);
      L = L * __expf(M - mn) + lk * __expf(mk - mn);
      M = mn;
    }
    colmax[n * 64 + c] = M;
    colsum[n * 64 + c] = L;
  }
}

// ---------------- kernel 4: gate + final GEMM + residual ----------------
// out[n][c'][i] = gamma*( sum_c Wv[c'][c]*(Pm[i][c]*att[i][c]) + bv[c'] ) + x[n][c'][i]
__global__ void __launch_bounds__(128) k_out(
    const float* __restrict__ PmT, const float* __restrict__ O,
    const float* __restrict__ colmax, const float* __restrict__ colsum,
    const u16* __restrict__ Wvb, const float* __restrict__ bv,
    const float* __restrict__ gamma, const float* __restrict__ x,
    float* __restrict__ out) {
  int n = blockIdx.y;
  int lane = threadIdx.x & 31, wave = threadIdx.x >> 5;
  int h = lane >> 4, ln = lane & 15;
  int sb = blockIdx.x * 16;

  const float* Pm = PmT + (size_t)n * HWs * OUTc;
  const float* Ob = O + (size_t)n * HWs * OUTc;
  const float* cm = colmax + n * 64;
  const float* cs = colsum + n * 64;

  int i = sb + ln;
  const float* pr = Pm + (size_t)i * OUTc;
  const float* orow = Ob + (size_t)i * OUTc;

  Frag a0, a1;
#pragma unroll
  for (int e = 0; e < 16; ++e) {
    int k = (e < 8 ? e : e + 8) + 8 * h;
    int c0 = k, c1 = 32 + k;
    a0.u[e] = f2bf(pr[c0] * (__expf(orow[c0] - cm[c0]) / cs[c0]));
    a1.u[e] = f2bf(pr[c1] * (__expf(orow[c1] - cm[c1]) / cs[c1]));
  }

  float g = gamma[0];
#pragma unroll
  for (int ct = 0; ct < 4; ++ct) {
    int cp = (wave * 4 + ct) * 16 + ln;  // c'
    const u16* wrow = Wvb + (size_t)cp * OUTc;
    v8f acc = {};
    acc = wmma_bf16(a0.v, load_b(wrow, 0, h), acc);
    acc = wmma_bf16(a1.v, load_b(wrow, 32, h), acc);
    float bvc = bv[cp];
    size_t base = ((size_t)(n * Cc + cp)) * HWs + sb + 8 * h;
    const float* xr = x + base;
    float* orr = out + base;
#pragma unroll
    for (int v = 0; v < 8; ++v) orr[v] = g * (acc[v] + bvc) + xr[v];
  }
}

// ---------------- host launcher ----------------
extern "C" void kernel_launch(void* const* d_in, const int* in_sizes, int n_in,
                              void* d_out, int out_size, void* d_ws, size_t ws_size,
                              hipStream_t stream) {
  const float* x    = (const float*)d_in[0];
  const float* mask = (const float*)d_in[1];
  const float* Wf   = (const float*)d_in[2];
  const float* bf_  = (const float*)d_in[3];
  const float* Wg   = (const float*)d_in[4];
  const float* bg_  = (const float*)d_in[5];
  const float* Wh   = (const float*)d_in[6];
  const float* bh_  = (const float*)d_in[7];
  const float* Wm   = (const float*)d_in[8];
  const float* bm_  = (const float*)d_in[9];
  const float* Wv   = (const float*)d_in[10];
  const float* bv_  = (const float*)d_in[11];
  const float* gam  = (const float*)d_in[12];
  float* out = (float*)d_out;

  char* p = (char*)d_ws;
  auto alloc = [&](size_t bytes) -> char* {
    char* r = p;
    p += (bytes + 255) & ~(size_t)255;
    return r;
  };
  u16*   xT     = (u16*)alloc((size_t)Nn * HWs * Cc * 2);
  u16*   mT     = (u16*)alloc((size_t)Nn * HWs * Cc * 2);
  u16*   Wb     = (u16*)alloc((size_t)4 * OUTc * Cc * 2);
  u16*   Wvb    = (u16*)alloc((size_t)Cc * OUTc * 2);
  u16*   PfT    = (u16*)alloc((size_t)Nn * HWs * OUTc * 2);
  u16*   PgT    = (u16*)alloc((size_t)Nn * HWs * OUTc * 2);
  u16*   Ph     = (u16*)alloc((size_t)Nn * OUTc * HWs * 2);
  float* PmT    = (float*)alloc((size_t)Nn * HWs * OUTc * 4);
  float* O      = (float*)alloc((size_t)Nn * HWs * OUTc * 4);
  float* colmax = (float*)alloc((size_t)Nn * 64 * 4);
  float* colsum = (float*)alloc((size_t)Nn * 64 * 4);

  k_transpose_bf16<<<dim3(HWs / 32, Cc / 32, 2 * Nn), dim3(32, 8), 0, stream>>>(
      x, mask, xT, mT);
  k_pack_weights<<<(4 * OUTc * Cc + 255) / 256, 256, 0, stream>>>(
      Wf, Wg, Wh, Wm, Wv, Wb, Wvb);
  k_proj<<<dim3(HWs / 64, 4, Nn), 128, 0, stream>>>(
      xT, mT, Wb, bf_, bg_, bh_, bm_, PfT, PgT, Ph, PmT);
  k_attn<<<dim3(HWs / 64, Nn), 128, 0, stream>>>(PfT, PgT, Ph, O);
  k_colstats<<<Nn, 256, 0, stream>>>(O, colmax, colsum);
  k_out<<<dim3(HWs / 16, Nn), 128, 0, stream>>>(
      PmT, O, colmax, colsum, Wvb, bv_, gam, x, out);
}
